// SelfAttentionBlock_60017872995017
// MI455X (gfx1250) — matrix operand order repaired and checked by reading, not compiled, not run on hardware
//
#include <hip/hip_runtime.h>
#include <hip/hip_bf16.h>
#include <math.h>

// ---------------------------------------------------------------------------
// Problem constants (fixed by the reference)
// ---------------------------------------------------------------------------
#define BATCH 4
#define C_IN  512
#define NPIX  4096   // 64*64
#define C_QK  256
#define C_OUT 512

typedef __bf16 bf16_t;
typedef __attribute__((ext_vector_type(16))) __bf16 v16bf;
typedef __attribute__((ext_vector_type(8)))  __bf16 v8bf;
typedef __attribute__((ext_vector_type(8)))  float  v8f;
typedef __attribute__((ext_vector_type(4)))  float  v4f;

// Build a 16x32 bf16 A-fragment (contiguous k order) from a row pointer:
// lanes 0-15 hold K = {0..15}, lanes 16-31 hold K = {16..31}.
// rowPtr points at this lane's row, at the k-chunk base; grp = lane>>4.
__device__ inline v16bf load_a_frag(const bf16_t* rowPtr, int grp) {
    return *(const v16bf*)(rowPtr + grp * 16);
}

__device__ inline v8f wmma_bf16(v16bf a, v16bf b, v8f c) {
    return __builtin_amdgcn_wmma_f32_16x16x32_bf16(false, a, false, b, (short)0, c,
                                                   false, false);
}

// ---------------------------------------------------------------------------
// Pass 0a: transpose + convert f32 [B][512][4096] -> bf16 [B][4096][512]
// ---------------------------------------------------------------------------
__global__ __launch_bounds__(256) void tcvt_kernel(const float* __restrict__ in,
                                                   bf16_t* __restrict__ outT) {
    int b  = blockIdx.z;
    int n0 = blockIdx.x * 64;
    int c0 = blockIdx.y * 64;
    __shared__ float tile[64][65];
    int tx = threadIdx.x & 63;
    int ty = threadIdx.x >> 6;          // 0..3
    const float* ib = in + (size_t)b * C_IN * NPIX;
    for (int r = 0; r < 16; ++r) {
        int c = ty + r * 4;
        tile[c][tx] = ib[(size_t)(c0 + c) * NPIX + n0 + tx];
    }
    __syncthreads();
    bf16_t* ob = outT + (size_t)b * NPIX * C_IN;
    for (int r = 0; r < 16; ++r) {
        int n = ty + r * 4;
        ob[(size_t)(n0 + n) * C_IN + c0 + tx] = (bf16_t)tile[tx][n];
    }
}

// ---------------------------------------------------------------------------
// Pass 0b: elementwise f32 -> bf16 (weights)
// ---------------------------------------------------------------------------
__global__ void cvt_kernel(const float* __restrict__ in, bf16_t* __restrict__ out,
                           int nElem) {
    int i = blockIdx.x * blockDim.x + threadIdx.x;
    if (i < nElem) out[i] = (bf16_t)in[i];
}

// ---------------------------------------------------------------------------
// Pass 1: 1x1-conv projection as GEMM:  D[o][n] = sum_c W[o][c] * xT[n][c] + b[o]
//   A = W rows (bf16 [Cout][512]),  B = xT columns (bf16 [B][N][512])
//   mode 0: store (D*scl) as bf16 [B][N][Cout]   (for q, k)
//   mode 1: store (D*scl) as bf16 [B][Cout][N]   (for v, already transposed)
// Block: 8 waves; wave w owns o-subtile w (16 rows), all share 64 n columns.
// ---------------------------------------------------------------------------
__global__ __launch_bounds__(256) void proj_kernel(const bf16_t* __restrict__ Wb,
                                                   const float*  __restrict__ bias,
                                                   const bf16_t* __restrict__ xT,
                                                   bf16_t* __restrict__ outNT,
                                                   bf16_t* __restrict__ outT,
                                                   int Cout, float scl, int mode) {
    int b    = blockIdx.z;
    int n0   = blockIdx.x * 64;
    int o0   = blockIdx.y * 128;
    int w    = threadIdx.x >> 5;
    int lane = threadIdx.x & 31;
    int li   = lane & 15;
    int grp  = lane >> 4;
    int ot   = o0 + w * 16;                       // this wave's o-tile base

    const bf16_t* wrow = Wb + (size_t)(ot + li) * C_IN;
    const bf16_t* xb   = xT + (size_t)b * NPIX * C_IN;

    v8f acc[4];
    for (int j = 0; j < 4; ++j)
        for (int e = 0; e < 8; ++e) acc[j][e] = 0.f;

    for (int kc = 0; kc < C_IN / 32; ++kc) {
        v16bf a = load_a_frag(wrow + kc * 32, grp);
        for (int j = 0; j < 4; ++j) {
            const bf16_t* xp = xb + (size_t)(n0 + j * 16 + li) * C_IN + kc * 32 + grp * 16;
            v16bf bf = *(const v16bf*)xp;         // 32B contiguous B-fragment
            acc[j] = wmma_bf16(a, bf, acc[j]);
        }
    }

    // bias for rows o = ot + grp*8 + e
    v4f b0 = *(const v4f*)(bias + ot + grp * 8);
    v4f b1 = *(const v4f*)(bias + ot + grp * 8 + 4);
    float bb[8] = {b0[0], b0[1], b0[2], b0[3], b1[0], b1[1], b1[2], b1[3]};
    for (int j = 0; j < 4; ++j)
        for (int e = 0; e < 8; ++e) acc[j][e] = (acc[j][e] + bb[e]) * scl;

    if (mode == 0) {
        for (int j = 0; j < 4; ++j) {
            int n = n0 + j * 16 + li;
            v8bf pk;
            for (int e = 0; e < 8; ++e) pk[e] = (bf16_t)acc[j][e];
            *(v8bf*)(outNT + ((size_t)b * NPIX + n) * Cout + ot + grp * 8) = pk;
        }
    } else {
        for (int j = 0; j < 4; ++j) {
            int n = n0 + j * 16 + li;
            for (int e = 0; e < 8; ++e)
                outT[((size_t)b * Cout + ot + grp * 8 + e) * NPIX + n] = (bf16_t)acc[j][e];
        }
    }
}

// ---------------------------------------------------------------------------
// Pass 2: fused flash attention.
//   qs [B][N][C_QK] bf16 (pre-scaled by 1/16), ks [B][N][C_QK] bf16,
//   vs [B][C_OUT][N] bf16, out f32 [B][C_OUT][N].
// Block = 256 thr = 8 waves, 32 query rows (2 row-tiles), BC=64 keys per iter.
// Wave w: rt = w&1 (row tile), cq = w>>1 (S col-subtile == O channel group).
// ---------------------------------------------------------------------------
__global__ __launch_bounds__(256) void attn_kernel(const bf16_t* __restrict__ qs,
                                                   const bf16_t* __restrict__ ks,
                                                   const bf16_t* __restrict__ vs,
                                                   float* __restrict__ out) {
    int b    = blockIdx.y;
    int n0   = blockIdx.x * 32;
    int w    = threadIdx.x >> 5;
    int lane = threadIdx.x & 31;
    int li   = lane & 15;
    int grp  = lane >> 4;
    int rt   = w & 1;
    int cq   = w >> 1;

    __shared__ float lmax[2][4][16];
    __shared__ float lsum[2][4][16];
    __shared__ __align__(32) bf16_t Plds[2][16][64];

    // Preload Q A-fragments for this wave's 16 rows (whole C_QK = 8 k-chunks).
    const bf16_t* qrow = qs + ((size_t)b * NPIX + n0 + rt * 16 + li) * C_QK;
    v16bf qf[8];
    for (int kc = 0; kc < 8; ++kc) qf[kc] = load_a_frag(qrow + kc * 32, grp);

    v8f  O[8];
    float m8[8], l8[8];
    for (int j = 0; j < 8; ++j)
        for (int e = 0; e < 8; ++e) O[j][e] = 0.f;
    for (int e = 0; e < 8; ++e) { m8[e] = -INFINITY; l8[e] = 0.f; }

    const bf16_t* ksb = ks + (size_t)b * NPIX * C_QK;
    const bf16_t* vsb = vs + (size_t)b * C_OUT * NPIX;

    for (int kt = 0; kt < NPIX / 64; ++kt) {
        // ---- S = (Q/16) . K^T for this wave's 16x16 (rt, cq) subtile ----
        const bf16_t* kcol = ksb + (size_t)(kt * 64 + cq * 16 + li) * C_QK;
        if (kt + 1 < NPIX / 64)
            __builtin_prefetch(ksb + (size_t)((kt + 1) * 64 + cq * 16 + li) * C_QK, 0, 1);
        v8f s;
        for (int e = 0; e < 8; ++e) s[e] = 0.f;
        for (int kc = 0; kc < 8; ++kc) {
            v16bf bf = *(const v16bf*)(kcol + kc * 32 + grp * 16);
            s = wmma_bf16(qf[kc], bf, s);
        }

        // ---- per-row max within this subtile (reduce across 16-lane group) ----
        float rmax[8];
        for (int e = 0; e < 8; ++e) {
            float x = s[e];
            x = fmaxf(x, __shfl_xor(x, 1, 32));
            x = fmaxf(x, __shfl_xor(x, 2, 32));
            x = fmaxf(x, __shfl_xor(x, 4, 32));
            x = fmaxf(x, __shfl_xor(x, 8, 32));
            rmax[e] = x;
        }
        if (li == 0)
            for (int e = 0; e < 8; ++e) lmax[rt][cq][grp * 8 + e] = rmax[e];
        __syncthreads();

        // ---- online softmax: combine max over the 4 col-subtiles, rescale O ----
        float mnew[8], scl[8];
        for (int e = 0; e < 8; ++e) {
            int r = grp * 8 + e;
            float mn = m8[e];
            mn = fmaxf(mn, lmax[rt][0][r]);
            mn = fmaxf(mn, lmax[rt][1][r]);
            mn = fmaxf(mn, lmax[rt][2][r]);
            mn = fmaxf(mn, lmax[rt][3][r]);
            mnew[e] = mn;
            scl[e]  = __expf(m8[e] - mn);
            m8[e]   = mn;
            l8[e]  *= scl[e];
        }
        for (int j = 0; j < 8; ++j)
            for (int e = 0; e < 8; ++e) O[j][e] *= scl[e];

        // ---- P = exp(S - m), row sums, P -> LDS (bf16) ----
        float rs[8];
        for (int e = 0; e < 8; ++e) {
            float p = __expf(s[e] - mnew[e]);
            Plds[rt][grp * 8 + e][cq * 16 + li] = (bf16_t)p;
            float x = p;
            x += __shfl_xor(x, 1, 32);
            x += __shfl_xor(x, 2, 32);
            x += __shfl_xor(x, 4, 32);
            x += __shfl_xor(x, 8, 32);
            rs[e] = x;
        }
        if (li == 0)
            for (int e = 0; e < 8; ++e) lsum[rt][cq][grp * 8 + e] = rs[e];
        __syncthreads();

        for (int e = 0; e < 8; ++e) {
            int r = grp * 8 + e;
            l8[e] += lsum[rt][0][r] + lsum[rt][1][r] + lsum[rt][2][r] + lsum[rt][3][r];
        }

        // ---- O += P . V   (wave owns 128 out channels = 8 col-subtiles) ----
        for (int kc2 = 0; kc2 < 2; ++kc2) {
            v16bf af = load_a_frag(&Plds[rt][li][kc2 * 32], grp);
            int mbase = kt * 64 + kc2 * 32;
            for (int j = 0; j < 8; ++j) {
                int o = cq * 128 + j * 16 + li;
                v16bf bf = *(const v16bf*)(vsb + (size_t)o * NPIX + mbase + grp * 16);
                O[j] = wmma_bf16(af, bf, O[j]);
            }
        }
        __syncthreads();   // protect lmax/Plds for next iteration
    }

    // ---- epilogue: O / l, store f32 [B][C_OUT][N] ----
    for (int e = 0; e < 8; ++e) l8[e] = 1.f / l8[e];
    for (int j = 0; j < 8; ++j) {
        int o = cq * 128 + j * 16 + li;
        float* op = out + ((size_t)b * C_OUT + o) * NPIX + n0 + rt * 16 + grp * 8;
        v4f s0, s1;
        for (int e = 0; e < 4; ++e) { s0[e] = O[j][e] * l8[e]; s1[e] = O[j][e + 4] * l8[e + 4]; }
        *(v4f*)op       = s0;
        *(v4f*)(op + 4) = s1;
    }
}

// ---------------------------------------------------------------------------
// Host launcher
// ---------------------------------------------------------------------------
extern "C" void kernel_launch(void* const* d_in, const int* in_sizes, int n_in,
                              void* d_out, int out_size, void* d_ws, size_t ws_size,
                              hipStream_t stream) {
    const float* qfeat = (const float*)d_in[0];   // [4][512][4096]
    const float* kfeat = (const float*)d_in[1];
    const float* Wq    = (const float*)d_in[2];   // [256][512]
    const float* bq    = (const float*)d_in[3];
    const float* Wk    = (const float*)d_in[4];
    const float* bk    = (const float*)d_in[5];
    const float* Wv    = (const float*)d_in[6];   // [512][512]
    const float* bv    = (const float*)d_in[7];
    float* out         = (float*)d_out;

    char* ws = (char*)d_ws;
    bf16_t* xqT = (bf16_t*)(ws);                                   // 16 MB
    bf16_t* xkT = (bf16_t*)(ws + (16ull << 20));                   // 16 MB
    bf16_t* qsb = (bf16_t*)(ws + (32ull << 20));                   //  8 MB
    bf16_t* ksb = (bf16_t*)(ws + (40ull << 20));                   //  8 MB
    bf16_t* vsb = (bf16_t*)(ws + (48ull << 20));                   // 16 MB
    bf16_t* Wqb = (bf16_t*)(ws + (64ull << 20));                   // 256 KB
    bf16_t* Wkb = (bf16_t*)(ws + (64ull << 20) + (512u << 10));    // 256 KB
    bf16_t* Wvb = (bf16_t*)(ws + (64ull << 20) + (1u << 20));      // 512 KB

    // Pass 0: transpose+convert activations, convert weights.
    dim3 tg(NPIX / 64, C_IN / 64, BATCH);
    tcvt_kernel<<<tg, 256, 0, stream>>>(qfeat, xqT);
    tcvt_kernel<<<tg, 256, 0, stream>>>(kfeat, xkT);
    cvt_kernel<<<(C_QK * C_IN + 255) / 256, 256, 0, stream>>>(Wq, Wqb, C_QK * C_IN);
    cvt_kernel<<<(C_QK * C_IN + 255) / 256, 256, 0, stream>>>(Wk, Wkb, C_QK * C_IN);
    cvt_kernel<<<(C_OUT * C_IN + 255) / 256, 256, 0, stream>>>(Wv, Wvb, C_OUT * C_IN);

    // Pass 1: projections (q scaled by 1/sqrt(256) = 1/16).
    dim3 pgq(NPIX / 64, C_QK / 128, BATCH);
    proj_kernel<<<pgq, 256, 0, stream>>>(Wqb, bq, xqT, qsb, nullptr, C_QK, 1.0f / 16.0f, 0);
    proj_kernel<<<pgq, 256, 0, stream>>>(Wkb, bk, xkT, ksb, nullptr, C_QK, 1.0f, 0);
    dim3 pgv(NPIX / 64, C_OUT / 128, BATCH);
    proj_kernel<<<pgv, 256, 0, stream>>>(Wvb, bv, xkT, nullptr, vsb, C_OUT, 1.0f, 1);

    // Pass 2: fused flash attention.
    dim3 ag(NPIX / 32, BATCH);
    attn_kernel<<<ag, 256, 0, stream>>>(qsb, ksb, vsb, out);
}
